// HierarchicalMotionEncoder_55473797595855
// MI455X (gfx1250) — compile-verified, matrix operand order checked by reading, not run on hardware
//
#include <hip/hip_runtime.h>
#include <math.h>

#define MTOK   25600   // B*S
#define SEQ    200
#define DMODEL 512
#define NHEAD  8
#define NLAYER 6
#define DHEAD  64
#define DFFN   2048
#define NBATCH 128

typedef __attribute__((ext_vector_type(16))) _Float16 v16h;
typedef __attribute__((ext_vector_type(8)))  _Float16 v8h;
typedef __attribute__((ext_vector_type(8)))  float    v8f;

__device__ __forceinline__ float gelu_exact(float x) {
  return 0.5f * x * (1.0f + erff(x * 0.70710678118654752f));
}

enum { EP_BIAS = 0, EP_GELU = 1, EP_PE = 2 };

// weight pre-pass: out[N][K] f16 = transpose(in[K][N] f32)
__global__ void tconv_kernel(const float* __restrict__ in, _Float16* __restrict__ out,
                             int K, int N)
{
  const int idx = blockIdx.x * blockDim.x + threadIdx.x;
  if (idx >= K * N) return;
  const int n = idx / K, kk = idx % K;
  out[idx] = (_Float16)in[(size_t)kk * N + n];
}

template <int EP>
__device__ __forceinline__ void store_tile(float* __restrict__ C, const v8f& c,
                                           int mb, int nb, int N,
                                           const float* __restrict__ bias,
                                           const float* __restrict__ pe)
{
  const float b = bias[nb];
  #pragma unroll
  for (int r = 0; r < 8; ++r) {
    float vv = c[r] + b;
    if (EP == EP_GELU) vv = gelu_exact(vv);
    if (EP == EP_PE)   vv = vv * 22.6274169979695f + pe[((mb + r) % SEQ) * DMODEL + nb];
    C[(size_t)(mb + r) * N + nb] = vv;
  }
}

// C[M,N] = epilogue(A[M,K] @ Wt^T + bias[N]),  Wt is f16 [N][K] (pre-transposed)
// 256 threads = 8 waves; 128x64 macro tile; BK=64; each wave -> 32x32 strip (4 WMMA tiles).
template <int EP>
__global__ __launch_bounds__(256) void wmma_gemm_kernel(
    const float* __restrict__ A, const _Float16* __restrict__ Wt,
    const float* __restrict__ bias, float* __restrict__ C,
    int M, int N, int K, const float* __restrict__ pe)
{
  constexpr int LDH = 72;                 // halves per LDS row (keeps 16B alignment)
  __shared__ _Float16 As[128 * LDH];      // As[m][k]
  __shared__ _Float16 Bs[64 * LDH];       // Bs[n][k]

  const int bm    = blockIdx.y * 128;
  const int bn    = blockIdx.x * 64;
  const int t     = threadIdx.x;
  const int lane  = t & 31;
  const int wave  = t >> 5;
  const int wm    = wave >> 1;            // 0..3 -> 32-row subtile
  const int wn    = wave & 1;             // 0..1 -> 32-col subtile
  const int fr    = lane & 15;            // fragment row (A) / col (B)
  const int khalf = lane >> 4;

  // cooperative-load indices
  const int arow = t >> 1;                // 0..127
  const int acol = (t & 1) * 32;          // 0 / 32
  const int brow = t >> 2;                // 0..63  (n)
  const int bcol = (t & 3) * 16;          // 0/16/32/48 (k, in halves)

  v8f c00 = {}, c01 = {}, c10 = {}, c11 = {};

  for (int k0 = 0; k0 < K; k0 += 64) {
    // ---- stage A tile (f32 -> f16, row-major) ----
    const float* ap = A + (size_t)(bm + arow) * K + (k0 + acol);
    float4 f0 = *(const float4*)(ap + 0);
    float4 f1 = *(const float4*)(ap + 4);
    float4 f2 = *(const float4*)(ap + 8);
    float4 f3 = *(const float4*)(ap + 12);
    float4 f4 = *(const float4*)(ap + 16);
    float4 f5 = *(const float4*)(ap + 20);
    float4 f6 = *(const float4*)(ap + 24);
    float4 f7 = *(const float4*)(ap + 28);
    if (k0 + 64 < K) __builtin_prefetch(ap + 64, 0, 3);
    v8h p0, p1, p2, p3;
    p0[0]=(_Float16)f0.x; p0[1]=(_Float16)f0.y; p0[2]=(_Float16)f0.z; p0[3]=(_Float16)f0.w;
    p0[4]=(_Float16)f1.x; p0[5]=(_Float16)f1.y; p0[6]=(_Float16)f1.z; p0[7]=(_Float16)f1.w;
    p1[0]=(_Float16)f2.x; p1[1]=(_Float16)f2.y; p1[2]=(_Float16)f2.z; p1[3]=(_Float16)f2.w;
    p1[4]=(_Float16)f3.x; p1[5]=(_Float16)f3.y; p1[6]=(_Float16)f3.z; p1[7]=(_Float16)f3.w;
    p2[0]=(_Float16)f4.x; p2[1]=(_Float16)f4.y; p2[2]=(_Float16)f4.z; p2[3]=(_Float16)f4.w;
    p2[4]=(_Float16)f5.x; p2[5]=(_Float16)f5.y; p2[6]=(_Float16)f5.z; p2[7]=(_Float16)f5.w;
    p3[0]=(_Float16)f6.x; p3[1]=(_Float16)f6.y; p3[2]=(_Float16)f6.z; p3[3]=(_Float16)f6.w;
    p3[4]=(_Float16)f7.x; p3[5]=(_Float16)f7.y; p3[6]=(_Float16)f7.z; p3[7]=(_Float16)f7.w;
    _Float16* adst = &As[arow * LDH + acol];
    *(v8h*)(adst + 0)  = p0;
    *(v8h*)(adst + 8)  = p1;
    *(v8h*)(adst + 16) = p2;
    *(v8h*)(adst + 24) = p3;

    // ---- stage B tile (already f16 and [N][K]): pure b128 copies ----
    const _Float16* wp = Wt + (size_t)(bn + brow) * K + (k0 + bcol);
    v8h b0 = *(const v8h*)(wp + 0);
    v8h b1 = *(const v8h*)(wp + 8);
    if (k0 + 64 < K) __builtin_prefetch(wp + 64, 0, 3);
    _Float16* bdst = &Bs[brow * LDH + bcol];
    *(v8h*)(bdst + 0) = b0;
    *(v8h*)(bdst + 8) = b1;

    __syncthreads();

    #pragma unroll
    for (int ks = 0; ks < 2; ++ks) {
      // 16-bit operand layout: lanes 0-15 hold K {0..7,16..23}, lanes 16-31 {8..15,24..31}
      const int ko = ks * 32 + khalf * 8;
      const _Float16* a0p = &As[(wm * 32 + fr) * LDH + ko];
      const _Float16* a1p = a0p + 16 * LDH;
      const _Float16* b0p = &Bs[(wn * 32 + fr) * LDH + ko];
      const _Float16* b1p = b0p + 16 * LDH;
      v8h a0lo = *(const v8h*)a0p,        a0hi = *(const v8h*)(a0p + 16);
      v8h a1lo = *(const v8h*)a1p,        a1hi = *(const v8h*)(a1p + 16);
      v8h b0lo = *(const v8h*)b0p,        b0hi = *(const v8h*)(b0p + 16);
      v8h b1lo = *(const v8h*)b1p,        b1hi = *(const v8h*)(b1p + 16);
      v16h A0, A1, B0, B1;
      #pragma unroll
      for (int e = 0; e < 8; ++e) {
        A0[e] = a0lo[e]; A0[e + 8] = a0hi[e];
        A1[e] = a1lo[e]; A1[e + 8] = a1hi[e];
        B0[e] = b0lo[e]; B0[e + 8] = b0hi[e];
        B1[e] = b1lo[e]; B1[e + 8] = b1hi[e];
      }
      c00 = __builtin_amdgcn_wmma_f32_16x16x32_f16(false, A0, false, B0, (short)0, c00, false, false);
      c01 = __builtin_amdgcn_wmma_f32_16x16x32_f16(false, A0, false, B1, (short)0, c01, false, false);
      c10 = __builtin_amdgcn_wmma_f32_16x16x32_f16(false, A1, false, B0, (short)0, c10, false, false);
      c11 = __builtin_amdgcn_wmma_f32_16x16x32_f16(false, A1, false, B1, (short)0, c11, false, false);
    }
    __syncthreads();
  }

  // C/D layout: VGPR r -> M = r (+8 for lanes 16-31), N = lane&15
  const int mb0 = bm + wm * 32 + khalf * 8;
  const int mb1 = mb0 + 16;
  const int nb0 = bn + wn * 32 + fr;
  const int nb1 = nb0 + 16;
  store_tile<EP>(C, c00, mb0, nb0, N, bias, pe);
  store_tile<EP>(C, c01, mb0, nb1, N, bias, pe);
  store_tile<EP>(C, c10, mb1, nb0, N, bias, pe);
  store_tile<EP>(C, c11, mb1, nb1, N, bias, pe);
}

// Online-softmax attention: one wave per (b, h, q) row; 2 head-dims per lane.
__global__ __launch_bounds__(256) void attention_kernel(
    const float* __restrict__ q, const float* __restrict__ k,
    const float* __restrict__ v, const float* __restrict__ tb,
    float* __restrict__ ctx, int layer)
{
  const int lane = threadIdx.x & 31;
  const int wave = threadIdx.x >> 5;
  const int row  = blockIdx.x * 8 + wave;      // (b*H + h)*S + qi
  const int qi = row % SEQ;
  const int tt = row / SEQ;
  const int hh = tt % NHEAD;
  const int b  = tt / NHEAD;

  const size_t base = ((size_t)(b * SEQ + qi)) * DMODEL + hh * DHEAD + lane * 2;
  float2 qv = *(const float2*)(q + base);
  const float* tbp = tb + (((size_t)layer * NHEAD + hh) * SEQ + qi) * SEQ;

  float mmax = -1e30f, ssum = 0.f;
  float2 acc = make_float2(0.f, 0.f);
  const float inv_scale = 0.125f;              // 1/sqrt(64)

  for (int j = 0; j < SEQ; ++j) {
    const size_t bj = ((size_t)(b * SEQ + j)) * DMODEL + hh * DHEAD + lane * 2;
    float2 kv = *(const float2*)(k + bj);
    float p = qv.x * kv.x + qv.y * kv.y;
    #pragma unroll
    for (int off = 16; off; off >>= 1) p += __shfl_xor(p, off, 32);
    float s = p * inv_scale + tbp[j];
    float mn = fmaxf(mmax, s);
    float corr = __expf(mmax - mn);
    float w = __expf(s - mn);
    ssum = ssum * corr + w;
    float2 vv = *(const float2*)(v + bj);
    acc.x = acc.x * corr + w * vv.x;
    acc.y = acc.y * corr + w * vv.y;
    mmax = mn;
  }
  const float invs = 1.f / ssum;
  float2 o; o.x = acc.x * invs; o.y = acc.y * invs;
  *(float2*)(ctx + base) = o;
}

// h = LayerNorm(h + t) * g + be ; one block (256 thr) per token row of 512.
__global__ __launch_bounds__(256) void add_ln_kernel(
    float* __restrict__ h, const float* __restrict__ t,
    const float* __restrict__ g, const float* __restrict__ be)
{
  __shared__ float red[16];
  const int row = blockIdx.x;
  const int tid = threadIdx.x;
  float* hp = h + (size_t)row * DMODEL;
  const float* tp = t + (size_t)row * DMODEL;
  float x0 = hp[tid] + tp[tid];
  float x1 = hp[tid + 256] + tp[tid + 256];
  float s = x0 + x1, s2 = x0 * x0 + x1 * x1;
  #pragma unroll
  for (int off = 16; off; off >>= 1) {
    s  += __shfl_xor(s, off, 32);
    s2 += __shfl_xor(s2, off, 32);
  }
  if ((tid & 31) == 0) { red[tid >> 5] = s; red[8 + (tid >> 5)] = s2; }
  __syncthreads();
  float ts = 0.f, ts2 = 0.f;
  #pragma unroll
  for (int i = 0; i < 8; ++i) { ts += red[i]; ts2 += red[8 + i]; }
  const float mean = ts * (1.f / DMODEL);
  const float var  = ts2 * (1.f / DMODEL) - mean * mean;
  const float rstd = rsqrtf(var + 1e-5f);
  hp[tid]       = (x0 - mean) * rstd * g[tid] + be[tid];
  hp[tid + 256] = (x1 - mean) * rstd * g[tid + 256] + be[tid + 256];
}

// local pooling (50 bins of 4 then mean) == global mean over S; comb = [glob, glob]
__global__ void pool_kernel(const float* __restrict__ h, float* __restrict__ comb)
{
  const int idx = blockIdx.x * blockDim.x + threadIdx.x;  // 0..B*D-1
  if (idx >= NBATCH * DMODEL) return;
  const int b = idx / DMODEL, d = idx % DMODEL;
  const float* p = h + (size_t)b * SEQ * DMODEL + d;
  float s = 0.f;
  for (int j = 0; j < SEQ; ++j) s += p[(size_t)j * DMODEL];
  s *= (1.f / SEQ);
  comb[(size_t)b * 1024 + d]       = s;
  comb[(size_t)b * 1024 + 512 + d] = s;
}

// small second head layer with output split (style/skill halves)
__global__ void head2_kernel(const float* __restrict__ hid, const float* __restrict__ W2,
                             const float* __restrict__ b2, float* __restrict__ out_lo,
                             float* __restrict__ out_hi, int N, int K, int split)
{
  const int idx = blockIdx.x * blockDim.x + threadIdx.x;
  if (idx >= NBATCH * N) return;
  const int b = idx / N, n = idx % N;
  const float* hp = hid + (size_t)b * K;
  float acc = b2[n];
  for (int kk = 0; kk < K; ++kk) acc += hp[kk] * W2[(size_t)kk * N + n];
  if (n < split) out_lo[b * split + n] = acc;
  else           out_hi[b * (N - split) + (n - split)] = acc;
}

extern "C" void kernel_launch(void* const* d_in, const int* in_sizes, int n_in,
                              void* d_out, int out_size, void* d_ws, size_t ws_size,
                              hipStream_t stream) {
  const float* x   = (const float*)d_in[0];
  const float* Wp  = (const float*)d_in[1];
  const float* bp  = (const float*)d_in[2];
  const float* pe  = (const float*)d_in[3];
  const float* tb  = (const float*)d_in[4];
  const float* Wq  = (const float*)d_in[5];
  const float* bq  = (const float*)d_in[6];
  const float* Wk  = (const float*)d_in[7];
  const float* bk  = (const float*)d_in[8];
  const float* Wv  = (const float*)d_in[9];
  const float* bv  = (const float*)d_in[10];
  const float* Wo  = (const float*)d_in[11];
  const float* bo  = (const float*)d_in[12];
  const float* g1  = (const float*)d_in[13];
  const float* be1 = (const float*)d_in[14];
  const float* g2  = (const float*)d_in[15];
  const float* be2 = (const float*)d_in[16];
  const float* Wf1 = (const float*)d_in[17];
  const float* bf1 = (const float*)d_in[18];
  const float* Wf2 = (const float*)d_in[19];
  const float* bf2 = (const float*)d_in[20];
  const float* Ws1 = (const float*)d_in[21];
  const float* bs1 = (const float*)d_in[22];
  const float* Ws2 = (const float*)d_in[23];
  const float* bs2 = (const float*)d_in[24];
  const float* Wk1 = (const float*)d_in[25];
  const float* bk1 = (const float*)d_in[26];
  const float* Wk2 = (const float*)d_in[27];
  const float* bk2 = (const float*)d_in[28];
  const float* Wg1 = (const float*)d_in[29];
  const float* bg1 = (const float*)d_in[30];
  const float* Wg2 = (const float*)d_in[31];
  const float* bg2 = (const float*)d_in[32];

  // ---- workspace layout: f32 activations, then f16 transposed weights ----
  float* ws   = (float*)d_ws;
  float* h    = ws;
  float* tmp  = h   + (size_t)MTOK * DMODEL;
  float* hq   = tmp + (size_t)MTOK * DMODEL;
  float* hk   = hq  + (size_t)MTOK * DMODEL;
  float* hv   = hk  + (size_t)MTOK * DMODEL;
  float* ctx  = hv  + (size_t)MTOK * DMODEL;
  float* t1   = ctx + (size_t)MTOK * DMODEL;
  float* comb = t1  + (size_t)MTOK * DFFN;
  float* hid  = comb + (size_t)NBATCH * 1024;
  float* fend = hid  + (size_t)NBATCH * 512;

  _Float16* wh   = (_Float16*)fend;
  _Float16* WpT  = wh;  wh += (size_t)512 * 64;
  _Float16* WqT  = wh;  wh += (size_t)NLAYER * DMODEL * DMODEL;
  _Float16* WkT  = wh;  wh += (size_t)NLAYER * DMODEL * DMODEL;
  _Float16* WvT  = wh;  wh += (size_t)NLAYER * DMODEL * DMODEL;
  _Float16* WoT  = wh;  wh += (size_t)NLAYER * DMODEL * DMODEL;
  _Float16* Wf1T = wh;  wh += (size_t)NLAYER * DMODEL * DFFN;
  _Float16* Wf2T = wh;  wh += (size_t)NLAYER * DFFN * DMODEL;
  _Float16* Ws1T = wh;  wh += (size_t)1024 * 512;
  _Float16* Wk1T = wh;  wh += (size_t)1024 * 512;
  _Float16* Wg1T = wh;  wh += (size_t)1024 * 256;

  const dim3 blk(256);
  auto tconv = [&](const float* in, _Float16* out, int K, int N) {
    tconv_kernel<<<dim3((K * N + 255) / 256), blk, 0, stream>>>(in, out, K, N);
  };

  // 0) weight pre-pass: f32 [K,N] -> f16 [N,K]
  tconv(Wp, WpT, 64, DMODEL);
  for (int l = 0; l < NLAYER; ++l) {
    const size_t wo = (size_t)l * DMODEL * DMODEL;
    const size_t wf = (size_t)l * DMODEL * DFFN;
    tconv(Wq + wo, WqT + wo, DMODEL, DMODEL);
    tconv(Wk + wo, WkT + wo, DMODEL, DMODEL);
    tconv(Wv + wo, WvT + wo, DMODEL, DMODEL);
    tconv(Wo + wo, WoT + wo, DMODEL, DMODEL);
    tconv(Wf1 + wf, Wf1T + wf, DMODEL, DFFN);
    tconv(Wf2 + wf, Wf2T + wf, DFFN, DMODEL);
  }
  tconv(Ws1, Ws1T, 1024, 512);
  tconv(Wk1, Wk1T, 1024, 512);
  tconv(Wg1, Wg1T, 1024, 256);

  const dim3 gD(DMODEL / 64, MTOK / 128);   // N=512 token GEMMs
  const dim3 gF(DFFN / 64, MTOK / 128);     // N=2048 token GEMM

  // 1) input projection + sqrt(D) scale + positional encoding
  wmma_gemm_kernel<EP_PE><<<gD, blk, 0, stream>>>(x, WpT, bp, h, MTOK, DMODEL, 64, pe);

  for (int l = 0; l < NLAYER; ++l) {
    const size_t wo  = (size_t)l * DMODEL * DMODEL;
    const size_t wf  = (size_t)l * DMODEL * DFFN;
    const size_t bo_ = (size_t)l * DMODEL;
    wmma_gemm_kernel<EP_BIAS><<<gD, blk, 0, stream>>>(h, WqT + wo, bq + bo_, hq, MTOK, DMODEL, DMODEL, nullptr);
    wmma_gemm_kernel<EP_BIAS><<<gD, blk, 0, stream>>>(h, WkT + wo, bk + bo_, hk, MTOK, DMODEL, DMODEL, nullptr);
    wmma_gemm_kernel<EP_BIAS><<<gD, blk, 0, stream>>>(h, WvT + wo, bv + bo_, hv, MTOK, DMODEL, DMODEL, nullptr);
    attention_kernel<<<dim3(NBATCH * NHEAD * SEQ / 8), blk, 0, stream>>>(hq, hk, hv, tb, ctx, l);
    wmma_gemm_kernel<EP_BIAS><<<gD, blk, 0, stream>>>(ctx, WoT + wo, bo + bo_, tmp, MTOK, DMODEL, DMODEL, nullptr);
    add_ln_kernel<<<dim3(MTOK), blk, 0, stream>>>(h, tmp, g1 + bo_, be1 + bo_);
    wmma_gemm_kernel<EP_GELU><<<gF, blk, 0, stream>>>(h, Wf1T + wf, bf1 + (size_t)l * DFFN, t1, MTOK, DFFN, DMODEL, nullptr);
    wmma_gemm_kernel<EP_BIAS><<<gD, blk, 0, stream>>>(t1, Wf2T + wf, bf2 + bo_, tmp, MTOK, DMODEL, DFFN, nullptr);
    add_ln_kernel<<<dim3(MTOK), blk, 0, stream>>>(h, tmp, g2 + bo_, be2 + bo_);
  }

  // 2) pooling -> comb [B, 1024]
  pool_kernel<<<dim3((NBATCH * DMODEL + 255) / 256), blk, 0, stream>>>(h, comb);

  float* out     = (float*)d_out;
  float* o_sty1  = out;                 // [128,32]
  float* o_sty2  = out + 4096;          // [128,32]
  float* o_sk1   = out + 8192;          // [128,16]
  float* o_sk2   = out + 10240;         // [128,16]
  float* o_sig   = out + 12288;         // [128,5]
  float* o_h     = out + 12928;         // [128,200,512]

  // 3) heads: hidden via WMMA GEMM (M=128), tiny second layer via VALU
  wmma_gemm_kernel<EP_GELU><<<dim3(512 / 64, 1), blk, 0, stream>>>(comb, Ws1T, bs1, hid, NBATCH, 512, 1024, nullptr);
  head2_kernel<<<dim3((NBATCH * 64 + 255) / 256), blk, 0, stream>>>(hid, Ws2, bs2, o_sty1, o_sty2, 64, 512, 32);

  wmma_gemm_kernel<EP_GELU><<<dim3(512 / 64, 1), blk, 0, stream>>>(comb, Wk1T, bk1, hid, NBATCH, 512, 1024, nullptr);
  head2_kernel<<<dim3((NBATCH * 32 + 255) / 256), blk, 0, stream>>>(hid, Wk2, bk2, o_sk1, o_sk2, 32, 512, 16);

  wmma_gemm_kernel<EP_GELU><<<dim3(256 / 64, 1), blk, 0, stream>>>(comb, Wg1T, bg1, hid, NBATCH, 256, 1024, nullptr);
  head2_kernel<<<dim3((NBATCH * 5 + 255) / 256), blk, 0, stream>>>(hid, Wg2, bg2, o_sig, o_sig, 5, 256, 5);

  // 4) final hidden states h -> tail of d_out
  hipMemcpyAsync(o_h, h, (size_t)MTOK * DMODEL * sizeof(float),
                 hipMemcpyDeviceToDevice, stream);
}
